// NGCN_5050881540398
// MI455X (gfx1250) — matrix operand order repaired and checked by reading, not compiled, not run on hardware
//
#include <hip/hip_runtime.h>

#define IN_F   512
#define MF     48
#define OUT_F  240
#define LDW    520   // padded K-stride (elements) for weights in LDS: 1040B rows -> conflict-free

typedef __attribute__((ext_vector_type(16))) __bf16 v16bf;
typedef __attribute__((ext_vector_type(8)))  float  v8f;
typedef __attribute__((ext_vector_type(4)))  float  v4f;   // native clang vector for nontemporal loads

union FragB { int4 q[2]; v16bf v; unsigned short s[16]; unsigned d[8]; };

__device__ __forceinline__ unsigned short f2bf_bits(float f) {
  union { float f; unsigned u; } x; x.f = f;
  unsigned r = x.u + 0x7FFFu + ((x.u >> 16) & 1u);   // round-to-nearest-even
  return (unsigned short)(r >> 16);
}

// Pack two f32 -> one dword of two bf16 (lo, hi). Use the HW packed convert if declared.
#if __has_builtin(__builtin_amdgcn_cvt_pk_bf16_f32)
__device__ __forceinline__ unsigned cvt2bf(float lo, float hi) {
  typedef __attribute__((ext_vector_type(2))) __bf16 v2bf;
  union { v2bf v; unsigned u; } c;
  c.v = __builtin_amdgcn_cvt_pk_bf16_f32(lo, hi);
  return c.u;
}
#else
__device__ __forceinline__ unsigned cvt2bf(float lo, float hi) {
  return (unsigned)f2bf_bits(lo) | ((unsigned)f2bf_bits(hi) << 16);
}
#endif

// ---------------------------------------------------------------------------
// Convert + transpose the 5 weight matrices: w_i[k][n] (f32, 512x48)
//   -> wT[i][n][k] (bf16 bits, 48x512), n-major so a B-fragment lane reads
//      two 8-bf16 (16B) runs, k0..k0+7 and k0+16..k0+23, per K-step.
// ---------------------------------------------------------------------------
__global__ void conv_w_kernel(const float* __restrict__ w0, const float* __restrict__ w1,
                              const float* __restrict__ w2, const float* __restrict__ w3,
                              const float* __restrict__ w4, unsigned short* __restrict__ wT) {
  int i = blockIdx.x * blockDim.x + threadIdx.x;
  const int TOT = 5 * MF * IN_F;
  if (i >= TOT) return;
  int wi  = i / (MF * IN_F);
  int rem = i - wi * (MF * IN_F);
  int n   = rem / IN_F;
  int k   = rem - n * IN_F;
  const float* w = (wi == 0) ? w0 : (wi == 1) ? w1 : (wi == 2) ? w2 : (wi == 3) ? w3 : w4;
  wT[i] = f2bf_bits(w[k * MF + n]);
}

// ---------------------------------------------------------------------------
// Broadcast biases into d_out (N x 240): out[n, 48*br + f] = b_br[f].
// ---------------------------------------------------------------------------
__global__ void init_out_kernel(float* __restrict__ out,
                                const float* __restrict__ b0, const float* __restrict__ b1,
                                const float* __restrict__ b2, const float* __restrict__ b3,
                                const float* __restrict__ b4, long long total) {
  long long i = (long long)blockIdx.x * blockDim.x + threadIdx.x;
  if (i >= total) return;
  int j  = (int)(i % OUT_F);
  int br = j / MF;
  int f  = j - br * MF;
  const float* b = (br == 0) ? b0 : (br == 1) ? b1 : (br == 2) ? b2 : (br == 3) ? b3 : b4;
  out[i] = b[f];
}

__global__ void zero_kernel(float* __restrict__ p, long long total) {
  long long i = (long long)blockIdx.x * blockDim.x + threadIdx.x;
  if (i < total) p[i] = 0.0f;
}

// ---------------------------------------------------------------------------
// Fused 5-way GEMM: h_i = x @ w_i  (i = 0..4), bf16 WMMA, f32 accumulate.
// One non-temporal pass over x (205 MB). Block = 512 threads = 16 waves;
// each wave owns a 16-row tile and 15 accumulators (5 weights x 3 col-tiles).
// Weights live in LDS (5*48*520*2 = 249,600 B of the 320 KB WGP pool).
// Per K-step: phase 1 loads all 15 B fragments into distinct register sets
// (one ds burst + one wait), phase 2 runs 15 back-to-back WMMAs -> no
// WAR-hazard v_nops between matrix ops.
// ---------------------------------------------------------------------------
__global__ __launch_bounds__(512) void gemm_xw5_kernel(
    const float* __restrict__ x, const unsigned short* __restrict__ wT,
    float* __restrict__ h, int ntiles, int nrows) {
  extern __shared__ unsigned short wlds[];   // [5][48][LDW]

  const int tid = threadIdx.x;
  // Cooperative copy wT -> LDS, 8 bf16 (one int4) per op. 512 % 8 == 0, LDW % 8 == 0.
  const int TOT8 = 5 * MF * IN_F / 8;        // 15360
  for (int p = tid; p < TOT8; p += 512) {
    int idx = p * 8;
    int wi  = idx / (MF * IN_F);
    int rem = idx - wi * (MF * IN_F);
    int n   = rem / IN_F;
    int k   = rem - n * IN_F;
    *(int4*)&wlds[(wi * MF + n) * LDW + k] = *(const int4*)&wT[idx];
  }
  __syncthreads();

  const int wave = tid >> 5;
  const int lane = tid & 31;
  const int tile = blockIdx.x * 16 + wave;
  if (tile >= ntiles) return;                // wave-uniform: EXEC stays all-ones for WMMA

  const int row  = (tile << 4) + (lane & 15);
  const int bk   = (lane >> 4) << 3;         // A-frag K sub-chunk: 0 or 8
  const int kk   = (lane >> 4) << 3;         // B-frag K sub-chunk: 0 or 8
  const float* xrow = x + (size_t)row * IN_F;
  const unsigned short* wbase = wlds + (lane & 15) * LDW + kk;

  v8f acc[15];
  const v8f vzero = {0.f, 0.f, 0.f, 0.f, 0.f, 0.f, 0.f, 0.f};
#pragma unroll
  for (int t = 0; t < 15; ++t) acc[t] = vzero;

  for (int kb = 0; kb < IN_F; kb += 32) {
    if (kb + 32 < IN_F) __builtin_prefetch(xrow + kb + 32, 0, 1);  // global_prefetch_b8

    // A fragment: 16-bit A 16x32 layout. Lane reads K = kb+bk..+7 and kb+16+bk..+7.
    // x is single-use: stream it non-temporally, keep L2 for the SpMM phase.
    v4f f0 = __builtin_nontemporal_load((const v4f*)(xrow + kb + bk));
    v4f f1 = __builtin_nontemporal_load((const v4f*)(xrow + kb + bk + 4));
    v4f g0 = __builtin_nontemporal_load((const v4f*)(xrow + kb + 16 + bk));
    v4f g1 = __builtin_nontemporal_load((const v4f*)(xrow + kb + 16 + bk + 4));
    FragB a;
    a.d[0] = cvt2bf(f0[0], f0[1]);
    a.d[1] = cvt2bf(f0[2], f0[3]);
    a.d[2] = cvt2bf(f1[0], f1[1]);
    a.d[3] = cvt2bf(f1[2], f1[3]);
    a.d[4] = cvt2bf(g0[0], g0[1]);
    a.d[5] = cvt2bf(g0[2], g0[3]);
    a.d[6] = cvt2bf(g1[0], g1[1]);
    a.d[7] = cvt2bf(g1[2], g1[3]);

    // Phase 1: burst-load all 15 B fragments (distinct register sets).
    FragB b[15];
#pragma unroll
    for (int wi = 0; wi < 5; ++wi) {
#pragma unroll
      for (int c = 0; c < 3; ++c) {
        const unsigned short* bp = wbase + (wi * MF + c * 16) * LDW + kb;
        b[wi * 3 + c].q[0] = *(const int4*)(bp);
        b[wi * 3 + c].q[1] = *(const int4*)(bp + 16);
      }
    }
    // Phase 2: 15 back-to-back WMMAs (disjoint D registers -> no hazard NOPs).
#pragma unroll
    for (int t = 0; t < 15; ++t) {
      acc[t] = __builtin_amdgcn_wmma_f32_16x16x32_bf16(
          false, a.v, false, b[t].v, (short)0, acc[t], false, false);
    }
  }

  // C/D layout: VGPR r -> row = 8*(lane>>4) + r, col = lane&15.
  const int rbase = (tile << 4) + ((lane >> 4) << 3);
  const int colL  = lane & 15;
#pragma unroll
  for (int wi = 0; wi < 5; ++wi) {
    float* hb = h + (size_t)wi * (size_t)nrows * MF;
#pragma unroll
    for (int c = 0; c < 3; ++c) {
      v8f av = acc[wi * 3 + c];
#pragma unroll
      for (int r = 0; r < 8; ++r)
        hb[(size_t)(rbase + r) * MF + c * 16 + colL] = av[r];
    }
  }
}

// ---------------------------------------------------------------------------
// SpMM scatter: out[dst, f] += val[e] * in[src, f].
// 16-lane edge groups, 3 features each (f, f+16, f+32): a wave covers exactly
// 2 edges, gathers are coalesced 64B bursts, atomics are contiguous hardware
// global_atomic_add_f32 into L2-resident (19.2 MB) accumulators.
// ---------------------------------------------------------------------------
__global__ void spmm_kernel(const int* __restrict__ src, const int* __restrict__ dst,
                            const float* __restrict__ val, const float* __restrict__ in,
                            float* __restrict__ out, int ldo, int E) {
  long long i = (long long)blockIdx.x * blockDim.x + threadIdx.x;
  long long total = (long long)E << 4;
  if (i >= total) return;
  int e = (int)(i >> 4);
  int f = (int)(i & 15);
  float v = val[e];
  int s = src[e];
  int d = dst[e];
  const float* ip = in  + (size_t)s * MF + f;
  float*       op = out + (size_t)d * ldo + f;
  unsafeAtomicAdd(op +  0, v * ip[ 0]);
  unsafeAtomicAdd(op + 16, v * ip[16]);
  unsafeAtomicAdd(op + 32, v * ip[32]);
}

// ---------------------------------------------------------------------------
extern "C" void kernel_launch(void* const* d_in, const int* in_sizes, int n_in,
                              void* d_out, int out_size, void* d_ws, size_t ws_size,
                              hipStream_t stream) {
  const float* x     = (const float*)d_in[0];
  const int*   aidx  = (const int*)d_in[1];
  const float* aval  = (const float*)d_in[2];
  const int*   s1idx = (const int*)d_in[3];
  const float* s1val = (const float*)d_in[4];
  const int*   s2idx = (const int*)d_in[5];
  const float* s2val = (const float*)d_in[6];
  // dict order: w0,b0,w1,b1,...
  const float* w0 = (const float*)d_in[7];  const float* b0 = (const float*)d_in[8];
  const float* w1 = (const float*)d_in[9];  const float* b1 = (const float*)d_in[10];
  const float* w2 = (const float*)d_in[11]; const float* b2 = (const float*)d_in[12];
  const float* w3 = (const float*)d_in[13]; const float* b3 = (const float*)d_in[14];
  const float* w4 = (const float*)d_in[15]; const float* b4 = (const float*)d_in[16];
  float* out = (float*)d_out;

  const int E = in_sizes[2];          // 1,600,000
  const int N = in_sizes[0] / IN_F;   // 100,000 (divisible by 16)

  // Workspace: h0..h4 (5*N*48 f32) | t1,tA,tB (3*N*48 f32) | wT (5*48*512 bf16)
  float* h  = (float*)d_ws;
  float* t1 = h  + (size_t)5 * N * MF;
  float* tA = t1 + (size_t)N * MF;
  float* tB = tA + (size_t)N * MF;
  unsigned short* wT = (unsigned short*)(tB + (size_t)N * MF);

  const int B256 = 256;
  // 1. weight convert/transpose to bf16
  conv_w_kernel<<<(5 * MF * IN_F + B256 - 1) / B256, B256, 0, stream>>>(w0, w1, w2, w3, w4, wT);
  // 2. bias broadcast into output
  long long outtot = (long long)N * OUT_F;
  init_out_kernel<<<(int)((outtot + B256 - 1) / B256), B256, 0, stream>>>(out, b0, b1, b2, b3, b4, outtot);
  // 3. zero intermediate SpMM accumulators (t1, tA, tB are contiguous)
  long long ztot = (long long)3 * N * MF;
  zero_kernel<<<(int)((ztot + B256 - 1) / B256), B256, 0, stream>>>(t1, ztot);
  // 4. fused 5-way bf16 WMMA GEMM: one pass over x (512 threads = 16 waves/block)
  int ntiles  = N / 16;
  int gblocks = (ntiles + 15) / 16;
  size_t ldsB = (size_t)5 * MF * LDW * sizeof(unsigned short);   // 249,600 B
  gemm_xw5_kernel<<<gblocks, 512, ldsB, stream>>>(x, wT, h, ntiles, N);
  // 5. SpMM chains (src = idx[0] = first E, dst = idx[1] = second E)
  int sblocks = (int)((((long long)E << 4) + B256 - 1) / B256);
  const int* asrc = aidx;      const int* adst = aidx + E;
  spmm_kernel<<<sblocks, B256, 0, stream>>>(asrc, adst, aval, h,                      out + 0,   OUT_F, E);
  spmm_kernel<<<sblocks, B256, 0, stream>>>(asrc, adst, aval, h + (size_t)1 * N * MF, t1,        MF,    E);
  spmm_kernel<<<sblocks, B256, 0, stream>>>(asrc, adst, aval, t1,                     out + 48,  OUT_F, E);
  spmm_kernel<<<sblocks, B256, 0, stream>>>(asrc, adst, aval, h + (size_t)2 * N * MF, tA,        MF,    E);
  spmm_kernel<<<sblocks, B256, 0, stream>>>(asrc, adst, aval, tA,                     tB,        MF,    E);
  spmm_kernel<<<sblocks, B256, 0, stream>>>(asrc, adst, aval, tB,                     out + 96,  OUT_F, E);
  spmm_kernel<<<sblocks, B256, 0, stream>>>(s1idx, s1idx + E, s1val, h + (size_t)3 * N * MF, out + 144, OUT_F, E);
  spmm_kernel<<<sblocks, B256, 0, stream>>>(s2idx, s2idx + E, s2val, h + (size_t)4 * N * MF, out + 192, OUT_F, E);
}
